// Model_73383811219858
// MI455X (gfx1250) — matrix operand order repaired and checked by reading, not compiled, not run on hardware
//
#include <hip/hip_runtime.h>
#include <math.h>

// ---- Problem constants (from reference) ----
#define BB       16
#define LL       512
#define NIN      7
#define NOUT     7
#define DMODEL   256
#define DINNER   512
#define DSTATE   16
#define DCONV    4
#define DTRANK   16
#define NLAYER   4
#define NPRED    96
#define NTOK     (BB * LL)          // 8192 tokens

#define LDS_STRIDE 40               // halves per padded LDS row (80B: 16B-aligned, bank-clean)

typedef __attribute__((ext_vector_type(16))) _Float16 v16h;
typedef __attribute__((ext_vector_type(8)))  _Float16 v8h;
typedef __attribute__((ext_vector_type(8)))  float    v8f;
typedef __attribute__((ext_vector_type(4)))  unsigned int u32x4;
typedef __attribute__((ext_vector_type(8)))  int      i32x8;
typedef __attribute__((ext_vector_type(4)))  int      i32x4;

#if defined(__gfx1250__) && __has_builtin(__builtin_amdgcn_tensor_load_to_lds)
#define USE_TDM 1
#else
#define USE_TDM 0
#endif

// ---------------------------------------------------------------------------
// TDM: stage a [rows x 32] f16 tile (row stride K elems) from global into LDS
// with padded rows (16 DW data + 4 DW pad = LDS_STRIDE halves). D# per ISA 8.3/8.4.
// ---------------------------------------------------------------------------
#if USE_TDM
__device__ __forceinline__ void tdm_stage_tile(const _Float16* gsrc,
                                               unsigned lds_off,
                                               int K, int rows, int dim1_total) {
    unsigned long long ga = (unsigned long long)(uintptr_t)gsrc;
    u32x4 g0 = {
        1u,                                               // count=1 (valid, user)
        lds_off,                                          // lds_addr
        (unsigned)(ga & 0xFFFFFFFFu),                     // global_addr[31:0]
        (unsigned)((ga >> 32) & 0x1FFFFFFu) | (2u << 30)  // global_addr[56:32] | type=2
    };
    i32x8 g1 = {
        (1 << 16) |                    // data_size = 2 bytes
        (1 << 20) |                    // pad_enable
        (3 << 22) |                    // pad_interval: 16 DWORDs
        (3 << 25),                     // pad_amount: 4 DWORDs
        (K & 0xFFFF) << 16,            // tensor_dim0[15:0]
        ((K >> 16) & 0xFFFF) | ((dim1_total & 0xFFFF) << 16), // dim0 hi | dim1 lo
        ((dim1_total >> 16) & 0xFFFF) | (32 << 16),           // dim1 hi | tile_dim0=32
        rows & 0xFFFF,                 // tile_dim1 (tile_dim2 = 0)
        K,                             // tensor_dim0_stride[31:0]
        0, 0
    };
    i32x4 gz  = {0, 0, 0, 0};
    i32x8 gz8 = {0, 0, 0, 0, 0, 0, 0, 0};
    __builtin_amdgcn_tensor_load_to_lds(g0, g1, gz, gz, gz8, 0);
}
#endif

// cooperative fallback: whole block copies [rows x 32] halves into padded LDS
__device__ __forceinline__ void coop_stage_tile(_Float16* lds, const _Float16* gsrc,
                                                int K, int rows, int tid, int nthreads) {
    for (int c = tid; c < rows * 4; c += nthreads) {      // 4 x 8-half chunks per row
        int r = c >> 2, ch = c & 3;
        *(v8h*)(lds + r * LDS_STRIDE + ch * 8) =
            *(const v8h*)(gsrc + (size_t)r * K + ch * 8);
    }
}

// ---------------------------------------------------------------------------
// Tiled WMMA GEMM:  C[M,N] = A[M,K] @ W[N,K]^T  (f16 operands, f32 accumulate)
// Block = 8 waves = 128M x (16*NT)N macro-tile. A/B K-chunks staged in
// double-buffered LDS (TDM when available), each wave does NT WMMAs per chunk.
// ---------------------------------------------------------------------------
template <int NT, bool RESID, bool W16OUT>
__global__ void gemm_wmma_kernel(const _Float16* __restrict__ A,
                                 const _Float16* __restrict__ W,
                                 float* __restrict__ C,
                                 _Float16* __restrict__ C16,
                                 int M, int N, int K) {
    __shared__ _Float16 ldsA[2][128 * LDS_STRIDE];
    __shared__ _Float16 ldsB[2][16 * NT * LDS_STRIDE];

    const int tid  = threadIdx.x;
    const int lane = tid & 31;
    const int wid  = tid >> 5;                  // 0..7
    const int nblk = N / (16 * NT);
    const int mb   = blockIdx.x / nblk;         // 128-row macro block
    const int nb   = blockIdx.x % nblk;
    const int row0 = mb * 128 + wid * 16;
    const int col0 = nb * 16 * NT;

    const _Float16* Ablk = A + (size_t)(mb * 128) * K;
    const _Float16* Wblk = W + (size_t)col0 * K;

#if USE_TDM
    if (wid == 0) {
        tdm_stage_tile(Ablk,      (unsigned)(uintptr_t)&ldsA[0][0], K, 128,     M);
        tdm_stage_tile(Wblk,      (unsigned)(uintptr_t)&ldsB[0][0], K, 16 * NT, N);
        if (K > 32) {
            tdm_stage_tile(Ablk + 32, (unsigned)(uintptr_t)&ldsA[1][0], K, 128,     M);
            tdm_stage_tile(Wblk + 32, (unsigned)(uintptr_t)&ldsB[1][0], K, 16 * NT, N);
        }
    }
#else
    coop_stage_tile(ldsA[0], Ablk, K, 128,     tid, blockDim.x);
    coop_stage_tile(ldsB[0], Wblk, K, 16 * NT, tid, blockDim.x);
    if (K > 32) {
        coop_stage_tile(ldsA[1], Ablk + 32, K, 128,     tid, blockDim.x);
        coop_stage_tile(ldsB[1], Wblk + 32, K, 16 * NT, tid, blockDim.x);
    }
#endif

    v8f acc[NT] = {};
    const int m     = lane & 15;
    const int half  = lane >> 4;

    for (int k0 = 0, it = 0; k0 < K; k0 += 32, ++it) {
        const int cur = it & 1;
#if USE_TDM
        if (wid == 0) {
            if (k0 + 32 < K) __builtin_amdgcn_s_wait_tensorcnt(2);
            else             __builtin_amdgcn_s_wait_tensorcnt(0);
        }
#endif
        __syncthreads();

        // A fragment: lane (16*half+m) holds K {8h..8h+7} and {8h+16..8h+23}
        const _Float16* la = &ldsA[cur][(wid * 16 + m) * LDS_STRIDE + half * 8];
        v8h a0 = *(const v8h*)(la);
        v8h a1 = *(const v8h*)(la + 16);
        v16h a;
#pragma unroll
        for (int e = 0; e < 8; ++e) { a[e] = a0[e]; a[8 + e] = a1[e]; }

#pragma unroll
        for (int i = 0; i < NT; ++i) {
            // B fragment: lane (16*half+n) holds K {8h..8h+7} and {8h+16..8h+23}
            const _Float16* lb = &ldsB[cur][(16 * i + m) * LDS_STRIDE + half * 8];
            v8h b0 = *(const v8h*)(lb);
            v8h b1 = *(const v8h*)(lb + 16);
            v16h b;
#pragma unroll
            for (int e = 0; e < 8; ++e) { b[e] = b0[e]; b[8 + e] = b1[e]; }
            acc[i] = __builtin_amdgcn_wmma_f32_16x16x32_f16(
                false, a, false, b, (short)0, acc[i], false, false);
        }

        __syncthreads();   // all reads of buf[cur] done before refill
        if (k0 + 64 < K) {
#if USE_TDM
            if (wid == 0) {
                tdm_stage_tile(Ablk + k0 + 64, (unsigned)(uintptr_t)&ldsA[cur][0], K, 128,     M);
                tdm_stage_tile(Wblk + k0 + 64, (unsigned)(uintptr_t)&ldsB[cur][0], K, 16 * NT, N);
            }
#else
            coop_stage_tile(ldsA[cur], Ablk + k0 + 64, K, 128,     tid, blockDim.x);
            coop_stage_tile(ldsB[cur], Wblk + k0 + 64, K, 16 * NT, tid, blockDim.x);
#endif
        }
    }

    // C/D layout: VGPR r -> M = r + 8*(lane>=16), N = lane&15
    const int mofs = half * 8;
#pragma unroll
    for (int i = 0; i < NT; ++i) {
#pragma unroll
        for (int r = 0; r < 8; ++r) {
            size_t idx = (size_t)(row0 + mofs + r) * N + (col0 + 16 * i + m);
            if (RESID) {
                float v = C[idx] + acc[i][r];
                C[idx] = v;
                if (W16OUT) C16[idx] = (_Float16)v;
            } else {
                C[idx] = acc[i][r];
                if (W16OUT) C16[idx] = (_Float16)acc[i][r];
            }
        }
    }
}

// ---------------------------------------------------------------------------
// Elementwise / small-K kernels
// ---------------------------------------------------------------------------
__device__ inline float silu_f(float x) { return x / (1.0f + __expf(-x)); }

__global__ void f32_to_f16_kernel(const float* __restrict__ src,
                                  _Float16* __restrict__ dst, int n) {
    int i = blockIdx.x * blockDim.x + threadIdx.x;
    if (i < n) dst[i] = (_Float16)src[i];
}

// h[t,m] = proj_b[m] + sum_{k<7} x[t,k]*proj_w[m,k]; also f16 copy for GEMM A
__global__ void input_proj_kernel(const float* __restrict__ x,
                                  const float* __restrict__ pw,
                                  const float* __restrict__ pb,
                                  float* __restrict__ h,
                                  _Float16* __restrict__ h16) {
    int idx = blockIdx.x * blockDim.x + threadIdx.x;
    if (idx >= NTOK * DMODEL) return;
    int m = idx & (DMODEL - 1);
    int t = idx >> 8;
    const float* xr = x + (size_t)t * NIN;
    const float* wr = pw + (size_t)m * NIN;
    float s = pb[m];
#pragma unroll
    for (int k = 0; k < NIN; ++k) s = fmaf(xr[k], wr[k], s);
    h[idx] = s;
    h16[idx] = (_Float16)s;
}

// causal depthwise conv (window 4) over L on x-half of xz, then SiLU
__global__ void conv_silu_kernel(const float* __restrict__ xz,   // [NTOK, 2*DINNER]
                                 const float* __restrict__ cw,   // [DINNER, 4]
                                 const float* __restrict__ cb,   // [DINNER]
                                 float* __restrict__ xs,         // [NTOK, DINNER]
                                 _Float16* __restrict__ xs16) {
    int idx = blockIdx.x * blockDim.x + threadIdx.x;
    if (idx >= NTOK * DINNER) return;
    int d = idx & (DINNER - 1);
    int t = idx >> 9;
    int l = t & (LL - 1);
    int b = t >> 9;
    float s = cb[d];
#pragma unroll
    for (int j = 0; j < DCONV; ++j) {
        int ls = l - (DCONV - 1) + j;
        if (ls >= 0)
            s = fmaf(cw[d * DCONV + j],
                     xz[(size_t)(b * LL + ls) * (2 * DINNER) + d], s);
    }
    float v = silu_f(s);
    xs[idx] = v;
    xs16[idx] = (_Float16)v;
}

// delta[t,d] = softplus( dtb[d] + sum_{r<16} xdbl[t,r] * dtw[d,r] )
__global__ void delta_kernel(const float* __restrict__ xdbl,  // [NTOK, 48]
                             const float* __restrict__ dtw,   // [DINNER, 16]
                             const float* __restrict__ dtb,
                             float* __restrict__ delta) {
    int idx = blockIdx.x * blockDim.x + threadIdx.x;
    if (idx >= NTOK * DINNER) return;
    int d = idx & (DINNER - 1);
    int t = idx >> 9;
    const float* r = xdbl + (size_t)t * (DTRANK + 2 * DSTATE);
    const float* w = dtw + (size_t)d * DTRANK;
    float s = dtb[d];
#pragma unroll
    for (int k = 0; k < DTRANK; ++k) s = fmaf(r[k], w[k], s);
    delta[idx] = (s > 20.0f) ? s : log1pf(__expf(s));
}

// selective scan: one lane per (b,d); 16 states in registers; B/C via LDS
__global__ void scan_kernel(const float* __restrict__ xs,     // u [NTOK, DINNER]
                            const float* __restrict__ delta,
                            const float* __restrict__ xdbl,
                            const float* __restrict__ A_log,  // [DINNER, 16]
                            const float* __restrict__ Dskip,
                            float* __restrict__ y) {
    __shared__ float sBC[2 * DSTATE];
    const int b = blockIdx.x >> 1;
    const int d = ((blockIdx.x & 1) << 8) + threadIdx.x;

    float Ad[DSTATE];
#pragma unroll
    for (int n = 0; n < DSTATE; ++n)
        Ad[n] = -__expf(A_log[(size_t)d * DSTATE + n]);

    float h[DSTATE] = {};
    const float dsk = Dskip[d];

    for (int l = 0; l < LL; ++l) {
        const size_t tok = (size_t)(b * LL + l);
        if (threadIdx.x < 2 * DSTATE)
            sBC[threadIdx.x] =
                xdbl[tok * (DTRANK + 2 * DSTATE) + DTRANK + threadIdx.x];
        __syncthreads();

        const size_t t = tok * DINNER + d;
        const float dlt = delta[t];
        const float u   = xs[t];
        const float du  = dlt * u;
        float acc = 0.0f;
#pragma unroll
        for (int n = 0; n < DSTATE; ++n) {
            float dA = __expf(dlt * Ad[n]);
            h[n] = fmaf(dA, h[n], du * sBC[n]);
            acc  = fmaf(h[n], sBC[DSTATE + n], acc);
        }
        y[t] = fmaf(u, dsk, acc);
        __syncthreads();
    }
}

// g16[t,d] = f16( y[t,d] * silu(z[t,d]) )
__global__ void gate_kernel(const float* __restrict__ y,
                            const float* __restrict__ xz,
                            _Float16* __restrict__ g16) {
    int idx = blockIdx.x * blockDim.x + threadIdx.x;
    if (idx >= NTOK * DINNER) return;
    int d = idx & (DINNER - 1);
    int t = idx >> 9;
    float z = xz[(size_t)t * (2 * DINNER) + DINNER + d];
    g16[idx] = (_Float16)(y[idx] * silu_f(z));
}

// out[b, t96, o] = sum_k h[b, 416+t96, k] * out_w[o, k]
__global__ void final_out_kernel(const float* __restrict__ h,
                                 const float* __restrict__ ow,
                                 float* __restrict__ out) {
    int idx = blockIdx.x * blockDim.x + threadIdx.x;
    if (idx >= BB * NPRED * NOUT) return;
    int o   = idx % NOUT;
    int t96 = (idx / NOUT) % NPRED;
    int b   = idx / (NOUT * NPRED);
    int l   = (LL - NPRED) + t96;
    const float* hr = h + (size_t)(b * LL + l) * DMODEL;
    const float* wr = ow + (size_t)o * DMODEL;
    float s = 0.0f;
#pragma unroll 8
    for (int k = 0; k < DMODEL; ++k) s = fmaf(hr[k], wr[k], s);
    out[idx] = s;
}

// ---------------------------------------------------------------------------
// Host orchestration
// ---------------------------------------------------------------------------
static inline int ceil_div(int a, int b) { return (a + b - 1) / b; }

extern "C" void kernel_launch(void* const* d_in, const int* in_sizes, int n_in,
                              void* d_out, int out_size, void* d_ws, size_t ws_size,
                              hipStream_t stream) {
    const float* x      = (const float*)d_in[0];
    const float* proj_w = (const float*)d_in[1];
    const float* proj_b = (const float*)d_in[2];
    const float* in_pw  = (const float*)d_in[3];   // [4,1024,256]
    const float* conv_w = (const float*)d_in[4];
    const float* conv_b = (const float*)d_in[5];
    const float* x_pw   = (const float*)d_in[6];   // [4,48,512]
    const float* dt_w   = (const float*)d_in[7];
    const float* dt_b   = (const float*)d_in[8];
    const float* A_log  = (const float*)d_in[9];
    const float* Dskip  = (const float*)d_in[10];
    const float* out_pw = (const float*)d_in[11];  // [4,256,512]
    const float* out_w  = (const float*)d_in[12];
    float* out = (float*)d_out;

    // ---- workspace layout ----
    const size_t n_iw = (size_t)NLAYER * 2 * DINNER * DMODEL;  // 1,048,576
    const size_t n_xw = (size_t)NLAYER * 48 * DINNER;          //    98,304
    const size_t n_ow = (size_t)NLAYER * DMODEL * DINNER;      //   524,288

    float* ws    = (float*)d_ws;
    float* h     = ws;                                         // [NTOK,256] f32
    float* xz    = h     + (size_t)NTOK * DMODEL;              // [NTOK,1024] f32
    float* xs    = xz    + (size_t)NTOK * 2 * DINNER;          // [NTOK,512] f32
    float* xdbl  = xs    + (size_t)NTOK * DINNER;              // [NTOK,48] f32
    float* delta = xdbl  + (size_t)NTOK * (DTRANK + 2 * DSTATE);
    float* ybuf  = delta + (size_t)NTOK * DINNER;              // [NTOK,512] f32
    _Float16* f16base = (_Float16*)(ybuf + (size_t)NTOK * DINNER);
    _Float16* h16  = f16base;                                  // [NTOK,256]
    _Float16* xs16 = h16  + (size_t)NTOK * DMODEL;             // [NTOK,512]
    _Float16* g16  = xs16 + (size_t)NTOK * DINNER;             // [NTOK,512]
    _Float16* iw16 = g16  + (size_t)NTOK * DINNER;
    _Float16* xw16 = iw16 + n_iw;
    _Float16* ow16 = xw16 + n_xw;

    const int TPB = 256;

    // one-time (per launch) weight conversion fp32 -> f16
    f32_to_f16_kernel<<<ceil_div((int)n_iw, TPB), TPB, 0, stream>>>(in_pw,  iw16, (int)n_iw);
    f32_to_f16_kernel<<<ceil_div((int)n_xw, TPB), TPB, 0, stream>>>(x_pw,   xw16, (int)n_xw);
    f32_to_f16_kernel<<<ceil_div((int)n_ow, TPB), TPB, 0, stream>>>(out_pw, ow16, (int)n_ow);

    // h = x @ proj_w.T + proj_b  (+ f16 copy)
    input_proj_kernel<<<ceil_div(NTOK * DMODEL, TPB), TPB, 0, stream>>>(
        x, proj_w, proj_b, h, h16);

    for (int i = 0; i < NLAYER; ++i) {
        const _Float16* iw = iw16 + (size_t)i * 2 * DINNER * DMODEL;
        const _Float16* xw = xw16 + (size_t)i * 48 * DINNER;
        const _Float16* ow = ow16 + (size_t)i * DMODEL * DINNER;
        const float* cw  = conv_w + (size_t)i * DINNER * DCONV;
        const float* cb  = conv_b + (size_t)i * DINNER;
        const float* dtw = dt_w   + (size_t)i * DINNER * DTRANK;
        const float* dtb = dt_b   + (size_t)i * DINNER;
        const float* Al  = A_log  + (size_t)i * DINNER * DSTATE;
        const float* Dk  = Dskip  + (size_t)i * DINNER;

        // xz = h @ in_w.T   M=8192, N=1024, K=256
        gemm_wmma_kernel<4, false, false>
            <<<(NTOK / 128) * ((2 * DINNER) / 64), TPB, 0, stream>>>(
                h16, iw, xz, nullptr, NTOK, 2 * DINNER, DMODEL);
        // conv + SiLU
        conv_silu_kernel<<<ceil_div(NTOK * DINNER, TPB), TPB, 0, stream>>>(
            xz, cw, cb, xs, xs16);
        // x_dbl = xs @ xw.T   M=8192, N=48, K=512
        gemm_wmma_kernel<3, false, false>
            <<<(NTOK / 128) * 1, TPB, 0, stream>>>(
                xs16, xw, xdbl, nullptr, NTOK, DTRANK + 2 * DSTATE, DINNER);
        // delta
        delta_kernel<<<ceil_div(NTOK * DINNER, TPB), TPB, 0, stream>>>(
            xdbl, dtw, dtb, delta);
        // selective scan
        scan_kernel<<<BB * (DINNER / 256), 256, 0, stream>>>(
            xs, delta, xdbl, Al, Dk, ybuf);
        // gated activation (f16)
        gate_kernel<<<ceil_div(NTOK * DINNER, TPB), TPB, 0, stream>>>(
            ybuf, xz, g16);
        // h += g @ ow.T  (residual, also refresh h16)   M=8192, N=256, K=512
        gemm_wmma_kernel<4, true, true>
            <<<(NTOK / 128) * (DMODEL / 64), TPB, 0, stream>>>(
                g16, ow, h, h16, NTOK, DMODEL, DINNER);
    }

    final_out_kernel<<<ceil_div(BB * NPRED * NOUT, TPB), TPB, 0, stream>>>(
        h, out_w, out);
    (void)in_sizes; (void)n_in; (void)out_size; (void)ws_size;
}
